// Attention_6133213298828
// MI455X (gfx1250) — matrix operand order repaired and checked by reading, not compiled, not run on hardware
//
#include <hip/hip_runtime.h>

#define DIMX 192
#define NH   3
#define DKH  64
#define VSEQ 17
#define QCOLS 576
#define XSTR 196   // padded LDS row stride for x / attn-out (floats)
#define QSTR 580   // padded LDS row stride for qkv (floats)

#define WQKV_FRAGS (36 * 6)  // 216 fragments of 512 bf16
#define WOUT_FRAGS (12 * 6)  //  72 fragments of 512 bf16

typedef __attribute__((ext_vector_type(16))) __bf16 v16bf;
typedef __attribute__((ext_vector_type(8)))  float  v8f;

// ---------------- weight pre-swizzle: fp32 row-major -> bf16 WMMA B-fragments --------------
// B (K=32 x N=16) fragment layout per wave32:
//   lane L, element i (0..15):  K = (L>=16 ? 8 : 0) + (i < 8 ? i : i + 8),  N = L & 15
__device__ __forceinline__ unsigned short f2bf(float f) {
  union { float f; unsigned int u; } in; in.f = f;
  unsigned int u = in.u;
  unsigned int r = (u + 0x7FFFu + ((u >> 16) & 1u)) >> 16;  // RNE
  return (unsigned short)r;
}

__global__ __launch_bounds__(256) void prep_weights(const float* __restrict__ Wqkv,
                                                    const float* __restrict__ Wout,
                                                    unsigned short* __restrict__ ws) {
  int gid = blockIdx.x * blockDim.x + threadIdx.x;
  const int total = (WQKV_FRAGS + WOUT_FRAGS) * 512;
  if (gid >= total) return;
  int f    = gid >> 9;
  int r    = gid & 511;
  int lane = r >> 4;
  int i    = r & 15;
  int Kl   = ((lane >> 4) << 3) + (i < 8 ? i : i + 8);   // K within 32-step
  int Nl   = lane & 15;
  float val;
  if (f < WQKV_FRAGS) {
    int nt = f / 6, ks = f % 6;
    val = Wqkv[(ks * 32 + Kl) * QCOLS + nt * 16 + Nl];
  } else {
    int f2 = f - WQKV_FRAGS;
    int nt = f2 / 6, ks = f2 % 6;
    val = Wout[(ks * 32 + Kl) * DIMX + nt * 16 + Nl];
  }
  ws[gid] = f2bf(val);
}

// ---------------- fragment helpers ----------------
__device__ __forceinline__ v16bf load_bfrag(const unsigned short* __restrict__ ws,
                                            int frag, int lane) {
  const uint4* p = (const uint4*)(ws + frag * 512 + lane * 16);
  union { uint4 u[2]; v16bf v; } un;
  un.u[0] = p[0];
  un.u[1] = p[1];
  return un.v;
}

__device__ __forceinline__ v16bf cvt16(float4 f0, float4 f1, float4 f2, float4 f3) {
  v16bf a;
  a[0]  = (__bf16)f0.x; a[1]  = (__bf16)f0.y; a[2]  = (__bf16)f0.z; a[3]  = (__bf16)f0.w;
  a[4]  = (__bf16)f1.x; a[5]  = (__bf16)f1.y; a[6]  = (__bf16)f1.z; a[7]  = (__bf16)f1.w;
  a[8]  = (__bf16)f2.x; a[9]  = (__bf16)f2.y; a[10] = (__bf16)f2.z; a[11] = (__bf16)f2.w;
  a[12] = (__bf16)f3.x; a[13] = (__bf16)f3.y; a[14] = (__bf16)f3.z; a[15] = (__bf16)f3.w;
  return a;
}

// A (M=16 x K=32) bf16 fragment, all 16 rows valid (rows 0-15).
//   lanes 0-15 : row = L,    elements -> K = {0..7, 16..23}  (+ks*32)
//   lanes 16-31: row = L-16, elements -> K = {8..15, 24..31} (+ks*32)
__device__ __forceinline__ v16bf build_afrag_full(const float* __restrict__ base, int stride,
                                                  int lane, int ks) {
  const float* p = base + (lane & 15) * stride + ks * 32 + ((lane >> 4) << 3);
  float4 f0 = *(const float4*)(p + 0);
  float4 f1 = *(const float4*)(p + 4);
  float4 f2 = *(const float4*)(p + 16);
  float4 f3 = *(const float4*)(p + 20);
  return cvt16(f0, f1, f2, f3);
}

// A fragment for the tail M-tile: only local row 0 (= global row 16) is real.
__device__ __forceinline__ v16bf build_afrag_row16(const float* __restrict__ base, int stride,
                                                   int lane, int ks) {
  v16bf a;
  if ((lane & 15) == 0) {
    const float* p = base + 16 * stride + ks * 32 + ((lane >> 4) << 3);
    float4 f0 = *(const float4*)(p + 0);
    float4 f1 = *(const float4*)(p + 4);
    float4 f2 = *(const float4*)(p + 16);
    float4 f3 = *(const float4*)(p + 20);
    a = cvt16(f0, f1, f2, f3);
  } else {
    #pragma unroll
    for (int i = 0; i < 16; ++i) a[i] = (__bf16)0.0f;
  }
  return a;
}

// ---------------- main kernel: one block per batch item ----------------
__global__ __launch_bounds__(192) void attn_rel_kernel(
    const float* __restrict__ x,
    const float* __restrict__ b_qkv,
    const float* __restrict__ key_rel,
    const float* __restrict__ key_rel_diag,
    const float* __restrict__ b_out,
    const unsigned short* __restrict__ wsW,
    float* __restrict__ out)
{
  __shared__ float sA[VSEQ * XSTR];             // x tile, later attn output (17x192)
  __shared__ float sQKV[VSEQ * QSTR];           // qkv result (17x576)
  __shared__ float sAttn[NH][VSEQ][VSEQ];       // softmaxed attention

  const int b    = blockIdx.x;
  const int tid  = threadIdx.x;
  const int lane = tid & 31;
  // wave-uniform values forced into SGPRs -> scalar branches below
  const int w    = __builtin_amdgcn_readfirstlane(tid >> 5);
  const int mt   = w & 1;        // M-tile (rows 0-15 / row 16)
  const int nb   = w >> 1;       // N-tile base for this wave

  // ---- stage x (17x192 fp32) into LDS via CDNA5 async direct-to-LDS path ----
  // NT hint: x is touched exactly once per kernel; don't let it evict weights from L2.
  {
    const float* xg = x + (size_t)b * VSEQ * DIMX;
    for (int i = tid; i < VSEQ * DIMX / 4; i += 192) {
      int e = i * 4;
      int row = e / DIMX, col = e % DIMX;
      unsigned lds_off = (unsigned)(unsigned long long)(&sA[row * XSTR + col]);
      const float* gp = xg + e;
      asm volatile("global_load_async_to_lds_b128 %0, %1, off th:TH_LOAD_NT"
                   :: "v"(lds_off), "v"(gp) : "memory");
    }
    asm volatile("s_wait_asynccnt 0x0" ::: "memory");
  }
  __syncthreads();

  // ---- QKV projection: (17x192) @ (192x576) via bf16 WMMA, fp32 accum ----
  {
    v16bf afr[6];
    if (mt == 0) {
      #pragma unroll
      for (int ks = 0; ks < 6; ++ks) afr[ks] = build_afrag_full(sA, XSTR, lane, ks);
    } else {
      #pragma unroll
      for (int ks = 0; ks < 6; ++ks) afr[ks] = build_afrag_row16(sA, XSTR, lane, ks);
    }
    for (int t = 0; t < 12; ++t) {
      int nt = nb + 3 * t;
      v8f c = {};
      #pragma unroll
      for (int ks = 0; ks < 6; ++ks) {
        v16bf bf = load_bfrag(wsW, nt * 6 + ks, lane);
        c = __builtin_amdgcn_wmma_f32_16x16x32_bf16(false, afr[ks], false, bf,
                                                    (short)0, c, false, false);
      }
      int col = nt * 16 + (lane & 15);
      float bq = b_qkv[col];
      if (mt == 0) {                       // scalar branch, unconditional stores
        #pragma unroll
        for (int r = 0; r < 8; ++r) {
          int row = r + ((lane >> 4) << 3);
          sQKV[row * QSTR + col] = c[r] + bq;
        }
      } else if (lane < 16) {              // only local row 0 == global row 16 is real
        sQKV[16 * QSTR + col] = c[0] + bq;
      }
    }
  }
  __syncthreads();

  // ---- attention logits + relative-position logits + softmax (exact fp32) ----
  // skew-trick collapsed: rel[p][q] = diag[p] (q==p) | off[p][q-1] (q>p) | off[p][q] (q<p)
  if (tid < NH * VSEQ) {
    const int h = tid / VSEQ, p = tid % VSEQ;
    float qreg[DKH];
    const float* qrow = &sQKV[p * QSTR + h * DKH];
    #pragma unroll
    for (int d = 0; d < DKH; ++d) qreg[d] = qrow[d];

    float logits[VSEQ];
    for (int j = 0; j < VSEQ; ++j) {
      const float* krow = &sQKV[j * QSTR + DIMX + h * DKH];
      float s = 0.f;
      #pragma unroll
      for (int d = 0; d < DKH; ++d) s += qreg[d] * krow[d];
      logits[j] = s;
    }
    const float* krel = key_rel + (size_t)p * 16 * DKH;
    for (int j = 0; j < 16; ++j) {
      float s = 0.f;
      #pragma unroll
      for (int d = 0; d < DKH; ++d) s += qreg[d] * krel[j * DKH + d];
      logits[j + (j >= p ? 1 : 0)] += s;
    }
    {
      float s = 0.f;
      #pragma unroll
      for (int d = 0; d < DKH; ++d) s += qreg[d] * key_rel_diag[d];
      logits[p] += s;
    }
    const float scale = 0.07216878364870323f;   // 192^-0.5
    float m = -3.4e38f;
    #pragma unroll
    for (int j = 0; j < VSEQ; ++j) { logits[j] *= scale; m = fmaxf(m, logits[j]); }
    float sum = 0.f;
    #pragma unroll
    for (int j = 0; j < VSEQ; ++j) { float e = __expf(logits[j] - m); logits[j] = e; sum += e; }
    float inv = 1.0f / sum;
    #pragma unroll
    for (int j = 0; j < VSEQ; ++j) sAttn[h][p][j] = logits[j] * inv;
  }
  __syncthreads();

  // ---- attn @ V -> sA (overlays x; x no longer needed) ----
  {
    const int h = tid / DKH;     // 0..2
    const int d = tid % DKH;
    float vcol[VSEQ];
    #pragma unroll
    for (int j = 0; j < VSEQ; ++j) vcol[j] = sQKV[j * QSTR + 2 * DIMX + h * DKH + d];
    for (int p = 0; p < VSEQ; ++p) {
      float acc = 0.f;
      #pragma unroll
      for (int j = 0; j < VSEQ; ++j) acc += sAttn[h][p][j] * vcol[j];
      sA[p * XSTR + h * DKH + d] = acc;
    }
  }
  __syncthreads();

  // ---- output projection: (17x192) @ (192x192) via bf16 WMMA ----
  {
    const unsigned short* wo = wsW + WQKV_FRAGS * 512;
    v16bf afr[6];
    if (mt == 0) {
      #pragma unroll
      for (int ks = 0; ks < 6; ++ks) afr[ks] = build_afrag_full(sA, XSTR, lane, ks);
    } else {
      #pragma unroll
      for (int ks = 0; ks < 6; ++ks) afr[ks] = build_afrag_row16(sA, XSTR, lane, ks);
    }
    float* og = out + (size_t)b * VSEQ * DIMX;
    for (int t = 0; t < 4; ++t) {
      int nt = nb + 3 * t;
      v8f c = {};
      #pragma unroll
      for (int ks = 0; ks < 6; ++ks) {
        v16bf bf = load_bfrag(wo, nt * 6 + ks, lane);
        c = __builtin_amdgcn_wmma_f32_16x16x32_bf16(false, afr[ks], false, bf,
                                                    (short)0, c, false, false);
      }
      int col = nt * 16 + (lane & 15);
      float bo = b_out[col];
      // NT stores: 'out' is written once and never re-read -> stream past L2.
      if (mt == 0) {
        #pragma unroll
        for (int r = 0; r < 8; ++r) {
          int row = r + ((lane >> 4) << 3);
          __builtin_nontemporal_store(c[r] + bo, &og[row * DIMX + col]);
        }
      } else if (lane < 16) {
        __builtin_nontemporal_store(c[0] + bo, &og[16 * DIMX + col]);
      }
    }
  }
}

extern "C" void kernel_launch(void* const* d_in, const int* in_sizes, int n_in,
                              void* d_out, int out_size, void* d_ws, size_t ws_size,
                              hipStream_t stream) {
  (void)n_in; (void)out_size; (void)ws_size;
  const float* x     = (const float*)d_in[0];
  const float* Wqkv  = (const float*)d_in[1];
  const float* bqkv  = (const float*)d_in[2];
  const float* krel  = (const float*)d_in[3];
  const float* kreld = (const float*)d_in[4];
  const float* Wout  = (const float*)d_in[5];
  const float* bout  = (const float*)d_in[6];
  float* outp        = (float*)d_out;
  unsigned short* ws = (unsigned short*)d_ws;

  const int nB = in_sizes[0] / (VSEQ * DIMX);

  const int prepTotal = (WQKV_FRAGS + WOUT_FRAGS) * 512;
  prep_weights<<<(prepTotal + 255) / 256, 256, 0, stream>>>(Wqkv, Wout, ws);
  attn_rel_kernel<<<nB, 192, 0, stream>>>(x, bqkv, krel, kreld, bout, ws, outp);
}
